// Encoder_Postnet_62904091017715
// MI455X (gfx1250) — matrix operand order repaired and checked by reading, not compiled, not run on hardware
//
#include <hip/hip_runtime.h>
#include <hip/hip_bf16.h>
#include <math.h>

// Problem constants (from reference): B=16, P=1024, T=8192, E=256
#define B_  16
#define P_  1024
#define T_  8192
#define E_  256
#define M_  (B_ * T_)          // 131072 GEMM rows
#define XLD 264                // padded LDS row stride (bf16 elems) to break bank conflicts

typedef __attribute__((ext_vector_type(16))) __bf16 v16bf;
typedef __attribute__((ext_vector_type(8)))  float  v8f;

// --- bf16 helpers done with integer bit ops (avoid __bf16 scalar arithmetic) ---
__device__ __forceinline__ unsigned short f32_to_bf16_rne(float f) {
    union { float f; unsigned int u; } c; c.f = f;
    unsigned int u = c.u;
    unsigned int r = u + 0x7FFFu + ((u >> 16) & 1u);   // round-to-nearest-even
    return (unsigned short)(r >> 16);
}
__device__ __forceinline__ float bf16_to_f32(unsigned short h) {
    union { unsigned int u; float f; } c; c.u = ((unsigned int)h) << 16;
    return c.f;
}

// ---------------------------------------------------------------------------
// Kernel 1: positional encoding table pe[T][E] (f32, lives in L2: 8 MB)
// ---------------------------------------------------------------------------
__global__ __launch_bounds__(256) void k_pe(float* __restrict__ pe) {
    long g = (long)blockIdx.x * 256 + threadIdx.x;   // < T*E
    int t = (int)(g >> 8);
    int e = (int)(g & (E_ - 1));
    int i = e >> 1;
    // div = exp(2i * (-ln(10000)/E))
    float ex   = (float)(2 * i) * (-9.210340371976184f / (float)E_);
    float freq = expf(ex);
    float ang  = (float)t * freq;
    pe[g] = (e & 1) ? cosf(ang) : sinf(ang);
}

// ---------------------------------------------------------------------------
// Kernel 2: split w_pos (f32 [K=E][N=E]) into bf16 hi/lo, pre-swizzled into
// WMMA B-fragment layout:  wf[ks][ntile][lane][slot]  (16 bf16 per lane)
//   lane<16 : n = ntile*16+lane, k = ks*32 + (slot<8 ? slot : 16+slot-8)
//   lane>=16: n = ntile*16+lane-16, k = ks*32 + 8 + (slot<8 ? slot : 16+slot-8)
// ---------------------------------------------------------------------------
__global__ __launch_bounds__(256) void k_wprep(const float* __restrict__ w_pos,
                                               unsigned short* __restrict__ wfhi,
                                               unsigned short* __restrict__ wflo) {
    int e    = blockIdx.x * 256 + threadIdx.x;       // 0 .. 65535
    int slot = e & 15;
    int lane = (e >> 4) & 31;
    int nt   = (e >> 9) & 15;
    int ks   = e >> 13;
    int n    = nt * 16 + (lane & 15);
    int k    = ks * 32 + ((lane >> 4) ? 8 : 0) + (slot & 7) + ((slot >> 3) ? 16 : 0);
    float v  = w_pos[k * E_ + n];
    unsigned short h = f32_to_bf16_rne(v);
    wfhi[e] = h;
    wflo[e] = f32_to_bf16_rne(v - bf16_to_f32(h));
}

// ---------------------------------------------------------------------------
// Kernel 3: aligner. One block per batch row. 256 chunks of 32 positions.
// Speculative chunk starts from a prefix sum of align-change boundaries
// (exact for valid duration expansions), then Jacobi fix-up to the true
// fixed point of the sequential recurrence.
// ---------------------------------------------------------------------------
__global__ __launch_bounds__(256) void k_align(const int* __restrict__ align_phone,
                                               const int* __restrict__ text_phone,
                                               int* __restrict__ idx_out) {
    __shared__ int s_text[P_];
    __shared__ int s_align[T_];
    __shared__ int s_pref[256];
    __shared__ int s_starts[257];
    __shared__ int s_ends[256];
    __shared__ int s_flag;

    const int b   = blockIdx.x;
    const int tid = threadIdx.x;
    const int* arow = align_phone + (long)b * T_;
    const int* trow = text_phone  + (long)b * P_;
    for (int j = tid; j < P_; j += 256) s_text[j]  = trow[j];
    for (int j = tid; j < T_; j += 256) s_align[j] = arow[j];
    __syncthreads();

    const int jb = tid * 32;
    int av[32];
#pragma unroll
    for (int j = 0; j < 32; ++j) av[j] = s_align[jb + j];

    // change-boundary count within this chunk
    int c = 0;
#pragma unroll
    for (int j = 0; j < 32; ++j) {
        int jj = jb + j;
        if (jj > 0) c += (s_align[jj] != s_align[jj - 1]) ? 1 : 0;
    }
    // block-wide inclusive scan (Hillis-Steele)
    s_pref[tid] = c; __syncthreads();
    for (int off = 1; off < 256; off <<= 1) {
        int v = (tid >= off) ? s_pref[tid - off] : 0;
        __syncthreads();
        s_pref[tid] += v;
        __syncthreads();
    }
    if (tid == 0) s_starts[0] = 0;
    s_starts[tid + 1] = min(s_pref[tid], P_ - 1);   // guess for chunk tid+1 start state
    __syncthreads();

    int* orow = idx_out + (long)b * T_;
    for (int pass = 0; pass < 257; ++pass) {
        int cur = s_starts[tid];
        int j0 = 0;
        if (tid == 0) { cur = 0; orow[0] = 0; j0 = 1; }   // idx_0 = 0 by definition
#pragma unroll
        for (int j = 0; j < 32; ++j) {
            if (j >= j0) {
                int a   = av[j];
                int nxt = min(cur + 1, P_ - 1);
                cur = (a == s_text[cur]) ? cur : nxt;
                orow[jb + j] = cur;
            }
        }
        s_ends[tid] = cur;
        if (tid == 0) s_flag = 0;
        __syncthreads();
        if (tid < 255 && s_starts[tid + 1] != s_ends[tid]) atomicOr(&s_flag, 1);
        __syncthreads();
        if (s_flag == 0) break;                        // this pass used correct starts
        s_starts[tid + 1] = s_ends[tid];               // Jacobi update
        __syncthreads();
    }
}

// ---------------------------------------------------------------------------
// Kernel 4: fused gather + PE + bf16x3 WMMA GEMM + full epilogue.
// Block tile: M=64 rows x N=256 cols. 8 waves:
//   wave w: m-subtile = w&3 (16 rows), n-half = w>>2 (8 n-tiles of 16 cols)
// X = aligned+PE staged in LDS as bf16 hi/lo in WMMA-A-friendly row-major.
// ---------------------------------------------------------------------------
__global__ __launch_bounds__(256) void k_main(
    const float* __restrict__ enc, const int* __restrict__ idx_ws,
    const float* __restrict__ pe,
    const unsigned short* __restrict__ wfhi, const unsigned short* __restrict__ wflo,
    const float* __restrict__ pitch, const int* __restrict__ beats,
    const float* __restrict__ w_pitch, const float* __restrict__ b_pitch,
    const float* __restrict__ emb_beats, const float* __restrict__ b_pos,
    float* __restrict__ out)
{
    extern __shared__ unsigned short smem[];
    unsigned short* xhi = smem;               // [64][XLD]
    unsigned short* xlo = smem + 64 * XLD;    // [64][XLD]

    const int  tid     = threadIdx.x;
    const long rowbase = (long)blockIdx.x * 64;

    // ---- stage X = gather(encoder_out, idx) + PE, split into bf16 hi/lo ----
    {
        int  i  = tid >> 2;                   // row 0..63
        int  q  = tid & 3;                    // 64-col quarter
        long r  = rowbase + i;
        int  t  = (int)(r & (T_ - 1));
        int  b  = (int)(r >> 13);
        int  id = idx_ws[r];
        const float* src = enc + ((long)b * P_ + id) * E_;
        const float* per = pe + (long)t * E_;
        int c0 = q * 64;
#pragma unroll
        for (int cc = 0; cc < 64; cc += 4) {
            float4 a = *(const float4*)(src + c0 + cc);
            float4 p = *(const float4*)(per + c0 + cc);
            float x[4] = { a.x + p.x, a.y + p.y, a.z + p.z, a.w + p.w };
#pragma unroll
            for (int j = 0; j < 4; ++j) {
                unsigned short h = f32_to_bf16_rne(x[j]);
                xhi[i * XLD + c0 + cc + j] = h;
                xlo[i * XLD + c0 + cc + j] = f32_to_bf16_rne(x[j] - bf16_to_f32(h));
            }
        }
    }
    __syncthreads();

    const int lane  = tid & 31;
    const int w     = tid >> 5;
    const int msub  = w & 3;     // which 16-row subtile
    const int nhalf = w >> 2;    // which 8 n-tiles

    v8f acc[8] = {};

    // A-fragment addressing (16-bit A 16x32 layout):
    //   lane<16 : M=lane,    K 0..7 in dw0-3, K 16..23 in dw4-7
    //   lane>=16: M=lane-16, K 8..15 in dw0-3, K 24..31 in dw4-7
    const int arow = msub * 16 + (lane & 15);
    const int koff = (lane >> 4) ? 8 : 0;

#pragma unroll
    for (int ks = 0; ks < 8; ++ks) {
        union { v16bf v; uint4 q[2]; } ahi, alo;
        int kb = ks * 32 + koff;
        ahi.q[0] = *(const uint4*)&xhi[arow * XLD + kb];
        ahi.q[1] = *(const uint4*)&xhi[arow * XLD + kb + 16];
        alo.q[0] = *(const uint4*)&xlo[arow * XLD + kb];
        alo.q[1] = *(const uint4*)&xlo[arow * XLD + kb + 16];
#pragma unroll
        for (int nti = 0; nti < 8; ++nti) {
            int nt = nhalf * 8 + nti;
            union { v16bf v; uint4 q[2]; } bhi, blo;
            const uint4* ph = (const uint4*)(wfhi + ((long)((ks * 16 + nt) * 32 + lane) * 16));
            const uint4* pl = (const uint4*)(wflo + ((long)((ks * 16 + nt) * 32 + lane) * 16));
            bhi.q[0] = ph[0]; bhi.q[1] = ph[1];
            blo.q[0] = pl[0]; blo.q[1] = pl[1];
            // bf16x3: hi*hi + hi*lo + lo*hi  (f32 accumulate)
            acc[nti] = __builtin_amdgcn_wmma_f32_16x16x32_bf16(
                false, ahi.v, false, bhi.v, (short)0, acc[nti], false, false);
            acc[nti] = __builtin_amdgcn_wmma_f32_16x16x32_bf16(
                false, ahi.v, false, blo.v, (short)0, acc[nti], false, false);
            acc[nti] = __builtin_amdgcn_wmma_f32_16x16x32_bf16(
                false, alo.v, false, bhi.v, (short)0, acc[nti], false, false);
        }
    }

    // ---- epilogue: out = aligned + pitch*w_pitch + b_pitch + emb_beats + (x@w + b_pos)
    // C layout: VGPR v -> row = v + 8*(lane>=16); col = lane&15
    const int ncol   = lane & 15;
    const int rowadd = (lane >> 4) * 8;
    float wp[8], bp[8], bps[8], e0[8], e1[8];
#pragma unroll
    for (int nti = 0; nti < 8; ++nti) {
        int n = (nhalf * 8 + nti) * 16 + ncol;
        wp[nti]  = w_pitch[n];
        bp[nti]  = b_pitch[n];
        bps[nti] = b_pos[n];
        e0[nti]  = emb_beats[n];
        e1[nti]  = emb_beats[E_ + n];
    }
#pragma unroll
    for (int v = 0; v < 8; ++v) {
        long r  = rowbase + msub * 16 + rowadd + v;
        int  b  = (int)(r >> 13);
        int  id = idx_ws[r];
        float pv = pitch[r];
        int   bt = beats[r];
        const float* ar = enc + ((long)b * P_ + id) * E_;
        float* orow = out + r * (long)E_;
#pragma unroll
        for (int nti = 0; nti < 8; ++nti) {
            int n = (nhalf * 8 + nti) * 16 + ncol;
            float aval = ar[n];
            float res  = aval + pv * wp[nti] + bp[nti] + (bt ? e1[nti] : e0[nti])
                       + acc[nti][v] + bps[nti];
            orow[n] = res;
        }
    }
}

// ---------------------------------------------------------------------------
extern "C" void kernel_launch(void* const* d_in, const int* in_sizes, int n_in,
                              void* d_out, int out_size, void* d_ws, size_t ws_size,
                              hipStream_t stream) {
    (void)in_sizes; (void)n_in; (void)out_size; (void)ws_size;
    const float* enc     = (const float*)d_in[0];
    const int*   align_p = (const int*)  d_in[1];
    const int*   text_p  = (const int*)  d_in[2];
    const float* pitch   = (const float*)d_in[3];
    const int*   beats   = (const int*)  d_in[4];
    const float* w_pitch = (const float*)d_in[5];
    const float* b_pitch = (const float*)d_in[6];
    const float* emb_b   = (const float*)d_in[7];
    const float* w_pos   = (const float*)d_in[8];
    const float* b_pos   = (const float*)d_in[9];
    float* out = (float*)d_out;

    // workspace partition
    char* ws = (char*)d_ws;
    int*            idx_ws = (int*)ws;                                    // 512 KB
    float*          pe_ws  = (float*)(ws + 524288);                      // 8 MB
    unsigned short* wfhi   = (unsigned short*)(ws + 524288 + 8388608);   // 128 KB
    unsigned short* wflo   = (unsigned short*)(ws + 524288 + 8388608 + 131072);

    k_pe   <<<(T_ * E_) / 256, 256, 0, stream>>>(pe_ws);
    k_wprep<<<65536 / 256,     256, 0, stream>>>(w_pos, wfhi, wflo);
    k_align<<<B_,              256, 0, stream>>>(align_p, text_p, idx_ws);

    size_t lds_bytes = (size_t)64 * XLD * 2 * sizeof(unsigned short);    // ~66 KB
    k_main <<<M_ / 64, 256, lds_bytes, stream>>>(enc, idx_ws, pe_ws, wfhi, wflo,
                                                 pitch, beats, w_pitch, b_pitch,
                                                 emb_b, b_pos, out);
}
